// LSTMEncoder_89266600280119
// MI455X (gfx1250) — compile-verified, matrix operand order checked by reading, not run on hardware
//

#include <hip/hip_runtime.h>
#include <hip/hip_bf16.h>

// Problem constants (match reference)
#define Vv   32000
#define Dd   512
#define Hh   512
#define Bb   64
#define Tt   512
#define G4H  2048     // 4*H
#define PAD_IDX 0
#define NBLK 32       // persistent recurrence: Hh/16 blocks

typedef __bf16 bf16_t;
typedef __attribute__((ext_vector_type(16))) __bf16 v16bf;
typedef __attribute__((ext_vector_type(8)))  float  v8f;

__device__ __forceinline__ bf16_t f2bf(float f) {
  unsigned u = __builtin_bit_cast(unsigned, f);
  u += 0x7FFFu + ((u >> 16) & 1u);          // round-to-nearest-even
  unsigned short h = (unsigned short)(u >> 16);
  return __builtin_bit_cast(bf16_t, h);
}

// Load a 16x32 (or 32x16) bf16 WMMA fragment from row-major [rows, ld] memory.
// Per ISA 7.12.2: lane L holds row (L&15); K elements {kb..kb+7} in elems 0-7
// and {kb+16..kb+23} in elems 8-15, kb = 8*(L>=16). A and B share this pattern.
// Works for global or LDS-backed pointers (inlined; addrspace inferred).
__device__ __forceinline__ v16bf load_frag(const bf16_t* __restrict__ p, int ld) {
  const int lane = threadIdx.x & 31;
  const int row  = lane & 15;
  const int kb   = (lane >> 4) << 3;   // 0 or 8
  const bf16_t* base = p + (size_t)row * ld + kb;
  union { uint4 u[2]; v16bf v; } f;
  f.u[0] = *(const uint4*)(base);        // K = kb .. kb+7
  f.u[1] = *(const uint4*)(base + 16);   // K = kb+16 .. kb+23
  return f.v;
}

// Device-wide barrier for a small co-resident grid (monotonic counter).
__device__ __forceinline__ void grid_barrier(int* ctr, int target) {
  __syncthreads();
  if (threadIdx.x == 0) {
    __threadfence();
    __hip_atomic_fetch_add(ctr, 1, __ATOMIC_RELEASE, __HIP_MEMORY_SCOPE_AGENT);
    while (__hip_atomic_load(ctr, __ATOMIC_ACQUIRE, __HIP_MEMORY_SCOPE_AGENT) < target) {
      __builtin_amdgcn_s_sleep(1);
    }
    __threadfence();
  }
  __syncthreads();
}

// ---------------------------------------------------------------------------
// Embedding gather + f32->bf16:  xe[tok, d] = (x[tok]==PAD ? 0 : emb[x[tok], d])
// ---------------------------------------------------------------------------
__global__ __launch_bounds__(256) void embed_kernel(const int* __restrict__ x,
                                                    const float* __restrict__ emb,
                                                    bf16_t* __restrict__ xe) {
  const int tok = blockIdx.x;              // 0 .. B*T-1
  const int d   = threadIdx.x * 2;         // 256 threads x 2 elems = 512
  const int xi  = x[tok];
  float2 e;
  if (xi == PAD_IDX) { e.x = 0.f; e.y = 0.f; }
  else               { e = *(const float2*)(emb + (size_t)xi * Dd + d); }
  unsigned pk = (unsigned)__builtin_bit_cast(unsigned short, f2bf(e.x))
              | ((unsigned)__builtin_bit_cast(unsigned short, f2bf(e.y)) << 16);
  *(unsigned*)(xe + (size_t)tok * Dd + d) = pk;
}

// ---------------------------------------------------------------------------
// Weight f32 -> bf16 conversion (same layout)
// ---------------------------------------------------------------------------
__global__ __launch_bounds__(256) void wcvt_kernel(const float* __restrict__ w,
                                                   bf16_t* __restrict__ wb, int n) {
  int i = blockIdx.x * blockDim.x + threadIdx.x;
  if (i < n) wb[i] = f2bf(w[i]);
}

// ---------------------------------------------------------------------------
// Zero-init recurrent state + grid-barrier counter
// ---------------------------------------------------------------------------
__global__ __launch_bounds__(256) void zero_state_kernel(bf16_t* __restrict__ ha,
                                                         bf16_t* __restrict__ hb,
                                                         int* __restrict__ ctr) {
  int i = blockIdx.x * blockDim.x + threadIdx.x;   // B*H = 32768 total
  bf16_t z = __builtin_bit_cast(bf16_t, (unsigned short)0);
  ha[i] = z; hb[i] = z;
  if (i == 0) *ctr = 0;
}

// ---------------------------------------------------------------------------
// Big input projection: xg[m, n] = sum_k A[m,k] * W[n,k]  + (b_ih[n] + b_hh[n])
// A: [M, K] bf16, W: [4H, K] bf16, xg: [M, 4H] f32.
// Block = 4 waves; block tile 128(M) x 64(N); each wave: 32(M) x 64(N)
// -> 8 WMMAs per K-slice against 2 A-frag + 4 B-frag loads.
// ---------------------------------------------------------------------------
__global__ __launch_bounds__(128) void gemm_xg_kernel(const bf16_t* __restrict__ A,
                                                      const bf16_t* __restrict__ W,
                                                      const float* __restrict__ b_ih,
                                                      const float* __restrict__ b_hh,
                                                      float* __restrict__ xg,
                                                      int K) {
  const int lane = threadIdx.x & 31;
  const int wave = threadIdx.x >> 5;         // 0..3
  const int n0 = blockIdx.x * 64;
  const int m0 = blockIdx.y * 128 + wave * 32;
  v8f acc[2][4] = {{{}, {}, {}, {}}, {{}, {}, {}, {}}};
  for (int kk = 0; kk < K; kk += 32) {
    v16bf a0 = load_frag(A + (size_t)m0 * K + kk, K);
    v16bf a1 = load_frag(A + (size_t)(m0 + 16) * K + kk, K);
#pragma unroll
    for (int j = 0; j < 4; ++j) {
      v16bf b = load_frag(W + (size_t)(n0 + 16 * j) * K + kk, K);
      acc[0][j] = __builtin_amdgcn_wmma_f32_16x16x32_bf16(
          false, a0, false, b, (short)0, acc[0][j], false, false);
      acc[1][j] = __builtin_amdgcn_wmma_f32_16x16x32_bf16(
          false, a1, false, b, (short)0, acc[1][j], false, false);
    }
  }
#pragma unroll
  for (int i = 0; i < 2; ++i) {
    const int mbase = m0 + 16 * i + ((lane >> 4) << 3);  // C row = r + 8*(lane>=16)
#pragma unroll
    for (int j = 0; j < 4; ++j) {
      const int n = n0 + 16 * j + (lane & 15);
      const float bias = b_ih[n] + b_hh[n];
#pragma unroll
      for (int r = 0; r < 8; ++r) {
        xg[(size_t)(mbase + r) * G4H + n] = acc[i][j][r] + bias;
      }
    }
  }
}

// ---------------------------------------------------------------------------
// Persistent LSTM recurrence: one launch runs all T steps of one layer.
// Grid: NBLK=32 blocks (16 h-cols each) x 128 threads (4 waves, 16 batch rows).
// W_hh slice (64 gate-rows x 512) staged into LDS once via async-to-LDS DMA;
// c kept in registers. Cross-step h handoff via bf16 ping-pong + grid barrier.
// Next step's xg slice prefetched (global_prefetch_b8) to hide HBM latency.
// ---------------------------------------------------------------------------
__global__ __launch_bounds__(128) void lstm_layer_kernel(const float* __restrict__ xg,   // [B*T, 4H]
                                                         const bf16_t* __restrict__ Whh, // [4H, H]
                                                         bf16_t* __restrict__ hA,        // [B, H] bf16 (zeroed)
                                                         bf16_t* __restrict__ hB,        // [B, H] bf16
                                                         float* __restrict__ hstate,     // [B, H] f32 (final)
                                                         float* __restrict__ cstate,     // [B, H] f32 (final)
                                                         bf16_t* __restrict__ yout,      // [B, T, H]
                                                         int store_y,
                                                         int* __restrict__ ctr) {
  __shared__ bf16_t sW[64 * Hh];   // 64 KB: rows r=0..63 -> gate (r>>4), col h0c+(r&15)

  const int lane = threadIdx.x & 31;
  const int wave = threadIdx.x >> 5;     // 0..3  -> batch rows 16*wave..
  const int h0c  = blockIdx.x * 16;      // this block's 16 h-columns
  const int m0   = wave * 16;

  // Stage this block's W_hh rows into LDS with async global->LDS DMA (ASYNCcnt).
  for (int i = threadIdx.x; i < 64 * (Hh / 8); i += blockDim.x) {
    const int r    = i / (Hh / 8);
    const int koff = (i % (Hh / 8)) * 8;
    const int grow = (r >> 4) * Hh + h0c + (r & 15);
    unsigned ldsa = (unsigned)(uintptr_t)(&sW[r * Hh + koff]);
    const bf16_t* gp = Whh + (size_t)grow * Hh + koff;
    asm volatile("global_load_async_to_lds_b128 %0, %1, off"
                 :: "v"(ldsa), "v"(gp) : "memory");
  }
  asm volatile("s_wait_asynccnt 0" ::: "memory");
  __syncthreads();

  const int hcol  = h0c + (lane & 15);
  const int bbase = m0 + ((lane >> 4) << 3);

  float creg[8];
#pragma unroll
  for (int r = 0; r < 8; ++r) creg[r] = 0.f;

  for (int t = 0; t < Tt; ++t) {
    const bf16_t* hin  = (t & 1) ? hB : hA;
    bf16_t*       hout = (t & 1) ? hA : hB;

    // Prefetch next step's xg slice; consumed after barrier + next K-loop.
    {
      const int tn = (t + 1 < Tt) ? (t + 1) : t;
#pragma unroll
      for (int r = 0; r < 8; ++r) {
        const float* xp = xg + ((size_t)(bbase + r) * Tt + tn) * G4H + hcol;
#pragma unroll
        for (int g = 0; g < 4; ++g) __builtin_prefetch(xp + g * Hh, 0, 3);
      }
    }

    v8f acc[4] = {{}, {}, {}, {}};       // gates i, f, g, o
    for (int kk = 0; kk < Hh; kk += 32) {
      v16bf a = load_frag(hin + (size_t)m0 * Hh + kk, Hh);
#pragma unroll
      for (int g = 0; g < 4; ++g) {
        v16bf b = load_frag(&sW[(g * 16) * Hh + kk], Hh);   // ds_load path
        acc[g] = __builtin_amdgcn_wmma_f32_16x16x32_bf16(
            false, a, false, b, (short)0, acc[g], false, false);
      }
    }

#pragma unroll
    for (int r = 0; r < 8; ++r) {
      const int b = bbase + r;
      const float* xr = xg + ((size_t)b * Tt + t) * G4H;
      const float gi = acc[0][r] + xr[0 * Hh + hcol];
      const float gf = acc[1][r] + xr[1 * Hh + hcol];
      const float gg = acc[2][r] + xr[2 * Hh + hcol];
      const float go = acc[3][r] + xr[3 * Hh + hcol];
      const float si = 1.f / (1.f + __expf(-gi));
      const float sf = 1.f / (1.f + __expf(-gf));
      const float so = 1.f / (1.f + __expf(-go));
      const float tg = tanhf(gg);
      const float c  = sf * creg[r] + si * tg;
      const float h  = so * tanhf(c);
      creg[r] = c;
      const int idx = b * Hh + hcol;
      hout[idx] = f2bf(h);
      if (store_y) yout[((size_t)b * Tt + t) * Hh + hcol] = f2bf(h);
      if (t == Tt - 1) {              // final state -> d_out
        hstate[idx] = h;
        cstate[idx] = c;
      }
    }

    if (t + 1 < Tt) grid_barrier(ctr, (t + 1) * NBLK);
  }
}

// ---------------------------------------------------------------------------
extern "C" void kernel_launch(void* const* d_in, const int* in_sizes, int n_in,
                              void* d_out, int out_size, void* d_ws, size_t ws_size,
                              hipStream_t stream) {
  (void)in_sizes; (void)n_in; (void)out_size; (void)ws_size;
  const int*   x    = (const int*)d_in[0];
  const float* emb  = (const float*)d_in[1];
  const float* Wih0 = (const float*)d_in[2];
  const float* Whh0 = (const float*)d_in[3];
  const float* bih0 = (const float*)d_in[4];
  const float* bhh0 = (const float*)d_in[5];
  const float* Wih1 = (const float*)d_in[6];
  const float* Whh1 = (const float*)d_in[7];
  const float* bih1 = (const float*)d_in[8];
  const float* bhh1 = (const float*)d_in[9];
  float* out = (float*)d_out;   // hn[2,B,H] then cn[2,B,H]

  float* h_l0 = out;                         // hn layer 0
  float* h_l1 = out + (size_t)Bb * Hh;       // hn layer 1
  float* c_l0 = out + 2 * (size_t)Bb * Hh;   // cn layer 0
  float* c_l1 = out + 3 * (size_t)Bb * Hh;   // cn layer 1

  // Workspace carve-up
  char* ws = (char*)d_ws;
  size_t off = 0;
  auto carve = [&](size_t bytes) -> char* {
    char* p = ws + off;
    off = (off + bytes + 255) & ~(size_t)255;
    return p;
  };
  const size_t MT = (size_t)Bb * Tt;                    // 32768 tokens
  bf16_t* xe   = (bf16_t*)carve(MT * Dd * sizeof(bf16_t));       // 32 MB
  bf16_t* y0   = (bf16_t*)carve(MT * Hh * sizeof(bf16_t));       // 32 MB
  float*  xg   = (float*)carve(MT * G4H * sizeof(float));        // 256 MB
  bf16_t* wih0 = (bf16_t*)carve((size_t)G4H * Dd * sizeof(bf16_t));
  bf16_t* whh0 = (bf16_t*)carve((size_t)G4H * Hh * sizeof(bf16_t));
  bf16_t* wih1 = (bf16_t*)carve((size_t)G4H * Hh * sizeof(bf16_t));
  bf16_t* whh1 = (bf16_t*)carve((size_t)G4H * Hh * sizeof(bf16_t));
  bf16_t* hbfA = (bf16_t*)carve((size_t)Bb * Hh * sizeof(bf16_t));
  bf16_t* hbfB = (bf16_t*)carve((size_t)Bb * Hh * sizeof(bf16_t));
  int*    ctr  = (int*)carve(256);

  const int wN = G4H * Dd;                 // 1048576 elems per weight matrix

  // 1) Embedding gather -> bf16
  embed_kernel<<<(int)MT, 256, 0, stream>>>(x, emb, xe);

  // 2) Weight conversions
  wcvt_kernel<<<wN / 256, 256, 0, stream>>>(Wih0, wih0, wN);
  wcvt_kernel<<<wN / 256, 256, 0, stream>>>(Whh0, whh0, wN);
  wcvt_kernel<<<wN / 256, 256, 0, stream>>>(Wih1, wih1, wN);
  wcvt_kernel<<<wN / 256, 256, 0, stream>>>(Whh1, whh1, wN);

  dim3 ggrid(G4H / 64, (unsigned)(MT / 128));   // (32, 256)

  // ---- Layer 0 ----
  gemm_xg_kernel<<<ggrid, 128, 0, stream>>>(xe, wih0, bih0, bhh0, xg, Dd);
  zero_state_kernel<<<(Bb * Hh) / 256, 256, 0, stream>>>(hbfA, hbfB, ctr);
  lstm_layer_kernel<<<NBLK, 128, 0, stream>>>(xg, whh0, hbfA, hbfB,
                                              h_l0, c_l0, y0, 1, ctr);

  // ---- Layer 1 ---- (input = y0; y output not needed)
  gemm_xg_kernel<<<ggrid, 128, 0, stream>>>(y0, wih1, bih1, bhh1, xg, Hh);
  zero_state_kernel<<<(Bb * Hh) / 256, 256, 0, stream>>>(hbfA, hbfB, ctr);
  lstm_layer_kernel<<<NBLK, 128, 0, stream>>>(xg, whh1, hbfA, hbfB,
                                              h_l1, c_l1, y0, 0, ctr);
}
